// VectorQuantizer_32100585571102
// MI455X (gfx1250) — matrix operand order repaired and checked by reading, not compiled, not run on hardware
//
#include <hip/hip_runtime.h>

typedef __attribute__((ext_vector_type(8)))  _Float16 v8h;
typedef __attribute__((ext_vector_type(16))) _Float16 v16h;
typedef __attribute__((ext_vector_type(8)))  float    v8f;

#define DDIM   256
#define KCODES 1024
#define NROWS  32768
#define TILE_BYTES 8192                    // 16 codes x 256 d x 2B
#define LOSS_SCALE (1.25f / 8388608.0f)    // (1+beta)/Ntot

// ---------------------------------------------------------------------------
// Prep: build [K][D] f32 transpose (gather source), [K][D] f16 copy (WMMA B),
// per-code squared norms, and zero the loss accumulator.
// ---------------------------------------------------------------------------
__global__ void vq_prep(const float* __restrict__ emb,   // [D][K] row-major
                        float* __restrict__ embT,        // [K][D]
                        _Float16* __restrict__ embH,     // [K][D]
                        float* __restrict__ enorm,       // [K]
                        float* __restrict__ loss) {
  int k = blockIdx.x * blockDim.x + threadIdx.x;
  if (k == 0) loss[0] = 0.0f;
  if (k >= KCODES) return;
  float nrm = 0.0f;
  for (int d = 0; d < DDIM; ++d) {
    float v = emb[(size_t)d * KCODES + k];   // coalesced across k
    nrm += v * v;
    embT[(size_t)k * DDIM + d] = v;
    embH[(size_t)k * DDIM + d] = (_Float16)v;
  }
  enorm[k] = nrm;
}

// Async DMA: 16 bytes per lane, global -> LDS, tracked by ASYNCcnt.
__device__ __forceinline__ void async_b128(unsigned lds_off, const void* gsrc) {
  unsigned long long ga = (unsigned long long)gsrc;
  asm volatile("global_load_async_to_lds_b128 %0, %1, off"
               :: "v"(lds_off), "v"(ga)
               : "memory");
}

// ---------------------------------------------------------------------------
// Main: one wave per 16-row tile of x; 8 waves/block share each 16-code
// B tile staged in LDS via async-to-LDS DMA, double buffered.
// dist = |e|^2 - 2*f.e accumulated entirely inside v_wmma_f32_16x16x32_f16
// (A pre-scaled by -2, C seeded with |e|^2 from an LDS-resident norm table).
// ---------------------------------------------------------------------------
__global__ __launch_bounds__(256)
void vq_main(const float* __restrict__ x,      // [NROWS][D]
             const float* __restrict__ embT,   // [K][D] f32
             const _Float16* __restrict__ embH,// [K][D] f16
             const float* __restrict__ enorm,  // [K]
             float* __restrict__ out,          // [NROWS][D]
             float* __restrict__ loss) {
  extern __shared__ _Float16 smem[];           // 2 x 8KB tiles + 4KB norms

  const int lane = threadIdx.x & 31;
  const int wave = threadIdx.x >> 5;
  const int tile = blockIdx.x * 8 + wave;      // 0..2047 row tiles
  const int m    = lane & 15;                  // matrix row / col within tile
  const int hiL  = lane >> 4;                  // lane half (0/1)

  // ---- stage |e|^2 table into LDS (visible after first loop barrier).
  float* senorm = (float*)(smem + TILE_BYTES); // halves: 2*8KB = 16KB offset
  for (int i = threadIdx.x; i < KCODES; i += 256) senorm[i] = enorm[i];

  // ---- A fragments (16 rows x 256 K), values = -2*x in f16, reg-resident.
  // Per ISA: lane half 0 holds K {0..7,16..23}+32c, half 1 holds {8..15,24..31}.
  v16h afrag[8];
  {
    const float* xrow = x + (size_t)(tile * 16 + m) * DDIM;
    const int kb = hiL * 8;
#pragma unroll
    for (int c = 0; c < 8; ++c) {
      const int d0 = c * 32 + kb;
      v8h lo, hi;
#pragma unroll
      for (int j = 0; j < 8; ++j) {
        lo[j] = (_Float16)(-2.0f * xrow[d0 + j]);
        hi[j] = (_Float16)(-2.0f * xrow[d0 + 16 + j]);
      }
      afrag[c] = __builtin_shufflevector(lo, hi, 0,1,2,3,4,5,6,7,8,9,10,11,12,13,14,15);
    }
  }

  float best[8];
  int   bidx[8];
#pragma unroll
  for (int i = 0; i < 8; ++i) { best[i] = 3.4e38f; bidx[i] = 0; }

  // ---- async staging setup: block loads one 8KB tile cooperatively,
  // wave w owns chunks (w*2,w*2+1) of 32 lanes x 16B each.
  const unsigned lds_base = (unsigned)(uintptr_t)(void*)smem;  // low 32b = LDS offset
  const char* embB = (const char*)embH;
  const unsigned c0 = ((wave * 2 + 0) * 32 + lane) * 16;       // byte offset in tile
  const unsigned c1 = ((wave * 2 + 1) * 32 + lane) * 16;

  // prologue: tile 0 -> buffer 0
  async_b128(lds_base + c0, embB + c0);
  async_b128(lds_base + c1, embB + c1);

  for (int t = 0; t < 64; ++t) {
    if (t < 63) {
      const char* src = embB + (size_t)(t + 1) * TILE_BYTES;
      unsigned    dst = lds_base + ((t + 1) & 1) * TILE_BYTES;
      async_b128(dst + c0, src + c0);
      async_b128(dst + c1, src + c1);
      asm volatile("s_wait_asynccnt 0x2" ::: "memory");  // tile t landed
    } else {
      asm volatile("s_wait_asynccnt 0x0" ::: "memory");
    }
    __syncthreads();                                     // tile t visible to block

    const int n = t * 16 + m;                 // code index for this lane's C column
    const _Float16* bbase = smem + (t & 1) * (TILE_BYTES / 2) + m * DDIM + hiL * 8;

    // stage all 8 B fragments into distinct registers: independent ds_loads
    // cluster, WMMA chain then issues without per-op full dscnt stalls.
    v16h bfrag[8];
#pragma unroll
    for (int c = 0; c < 8; ++c) {
      const v8h* bp = (const v8h*)(bbase + c * 32);      // ds_load_b128 x2
      bfrag[c] = __builtin_shufflevector(bp[0], bp[2],
                                         0,1,2,3,4,5,6,7,8,9,10,11,12,13,14,15);
    }

    const float en = senorm[n];                          // ds_load_b32
    v8f acc = { en, en, en, en, en, en, en, en };        // seed C with |e_n|^2
#pragma unroll
    for (int c = 0; c < 8; ++c) {
      acc = __builtin_amdgcn_wmma_f32_16x16x32_f16(
              /*neg_a=*/false, afrag[c], /*neg_b=*/false, bfrag[c],
              /*c_mod=*/(short)0, acc, /*reuse_a=*/false, /*reuse_b=*/false);
    }
    // acc[i] == dist(row i+8*hiL, code n); strict '<' keeps first occurrence.
#pragma unroll
    for (int i = 0; i < 8; ++i) {
      if (acc[i] < best[i]) { best[i] = acc[i]; bidx[i] = n; }
    }
    __syncthreads();    // all waves done reading buffer t&1 before DMA reuse
  }

  // ---- argmin across the 16 lanes of each half (columns n), tie -> smaller idx.
#pragma unroll
  for (int off = 1; off < 16; off <<= 1) {
#pragma unroll
    for (int i = 0; i < 8; ++i) {
      float ob = __shfl_xor(best[i], off, 32);
      int   oi = __shfl_xor(bidx[i], off, 32);
      if (ob < best[i] || (ob == best[i] && oi < bidx[i])) {
        best[i] = ob; bidx[i] = oi;
      }
    }
  }

  // Row r<8 lives in slots of lane half 0, r>=8 in half 1.
  int ridx[16];
#pragma unroll
  for (int r = 0; r < 16; ++r)
    ridx[r] = __shfl(bidx[r & 7], (r < 8) ? 0 : 16, 32);

  // ---- gather exact f32 codebook rows, write output, accumulate loss.
  float ls = 0.0f;
  const size_t obase = (size_t)tile * 16 * DDIM;
  for (int r = 0; r < 16; ++r) {
    const float* q    = embT + (size_t)ridx[r] * DDIM;
    const float* xr   = x + obase + (size_t)r * DDIM;
    float*       orow = out + obase + (size_t)r * DDIM;
#pragma unroll
    for (int c = lane; c < DDIM; c += 32) {
      float v = q[c];
      float d = v - xr[c];
      ls += d * d;
      orow[c] = v;
    }
  }
#pragma unroll
  for (int off = 16; off; off >>= 1) ls += __shfl_xor(ls, off, 32);
  if (lane == 0) atomicAdd(loss, ls * LOSS_SCALE);
}

// ---------------------------------------------------------------------------
extern "C" void kernel_launch(void* const* d_in, const int* in_sizes, int n_in,
                              void* d_out, int out_size, void* d_ws, size_t ws_size,
                              hipStream_t stream) {
  const float* x   = (const float*)d_in[0];   // [32,32,32,256] -> [32768][256]
  const float* emb = (const float*)d_in[1];   // [256][1024]

  float* out  = (float*)d_out;                     // quantized, 8388608 floats
  float* loss = out + (size_t)NROWS * DDIM;        // + scalar loss

  char* ws = (char*)d_ws;
  float*    embT  = (float*)ws;                                   // 1 MB
  _Float16* embH  = (_Float16*)(ws + (size_t)KCODES * DDIM * 4);  // 512 KB
  float*    enorm = (float*)(ws + (size_t)KCODES * DDIM * 6);     // 4 KB

  vq_prep<<<KCODES / 256, 256, 0, stream>>>(emb, embT, embH, enorm, loss);
  vq_main<<<NROWS / (16 * 8), 256, 2 * TILE_BYTES + KCODES * 4, stream>>>(
      x, embT, embH, enorm, out, loss);
}
